// StatePairwiseConcat_53824530154061
// MI455X (gfx1250) — matrix operand order repaired and checked by reading, not compiled, not run on hardware
//
#include <hip/hip_runtime.h>
#include <hip/hip_bf16.h>
#include <math.h>

typedef __attribute__((ext_vector_type(16))) _Float16 v16h;
typedef __attribute__((ext_vector_type(8)))  _Float16 v8h;
typedef __attribute__((ext_vector_type(8)))  float    v8f;

// Problem constants
#define BB 4
#define TT 1024
#define FF 500
#define FP 512        // F padded to multiple of 32
#define HH 128
#define PP 64
#define ROWS (BB*TT)  // 4096
#define KCTX (PP*HH)  // 8192
#define TCHUNK 64     // timesteps per k_ctx block (4 t-tiles of 16)
#define LROWS (TCHUNK + 63)   // 127 LDS rows: t0-63 .. t0+63

// ---------------------------------------------------------------------------
// 1) x [4096,500] f32 -> xh [4096,512] f16 (zero padded)
__global__ void k_convert_x(const float* __restrict__ x, _Float16* __restrict__ xh) {
    int idx = blockIdx.x * 256 + threadIdx.x;          // over 4096*512
    if (idx >= ROWS * FP) return;
    int row = idx >> 9;
    int f   = idx & (FP - 1);
    float v = (f < FF) ? x[(size_t)row * FF + f] : 0.0f;
    xh[idx] = (_Float16)v;
}

// ---------------------------------------------------------------------------
// 2) W_ih [384,500] f32 -> B-fragment packed f16, K=512, N=384 (24 n-tiles)
__global__ void k_pack_wih(const float* __restrict__ wih, _Float16* __restrict__ dst) {
    int idx = blockIdx.x * 256 + threadIdx.x;          // over 384*512
    if (idx >= 384 * FP) return;
    int g = idx >> 9;            // gate row 0..383  (N)
    int f = idx & (FP - 1);      // K
    float v = (f < FF) ? wih[(size_t)g * FF + f] : 0.0f;
    int ktile = f >> 5, kin = f & 31;
    int ntile = g >> 4, n   = g & 15;
    int lane = n + ((kin >= 16) ? 16 : 0);
    int j    = kin & 15;
    dst[((size_t)(ktile * 24 + ntile) * 32 + lane) * 16 + j] = (_Float16)v;
}

// ---------------------------------------------------------------------------
// 3) W_ctx [128, 16384] -> conv weights V[o][k], k = p*128 + h,
//    V = W_ctx[o, 8192 + h*64 + p]  (+ fold part1 row-sum into tap p==0),
//    stored B-fragment packed, K=8192, N=128 (8 n-tiles)
__global__ void k_pack_vctx(const float* __restrict__ wctx, _Float16* __restrict__ dst) {
    int idx = blockIdx.x * 256 + threadIdx.x;          // over 128*8192
    if (idx >= HH * KCTX) return;
    int o = idx >> 13;           // 0..127 (N)
    int k = idx & (KCTX - 1);    // 0..8191
    int p = k >> 7;
    int h = k & 127;
    float v = wctx[(size_t)o * (2 * HH * PP) + (HH + h) * PP + p];   // part2 tap
    if (p == 0) {
        float acc = 0.0f;
        const float* w1 = wctx + (size_t)o * (2 * HH * PP) + h * PP; // part1 row
        for (int q = 0; q < PP; ++q) acc += w1[q];
        v += acc;
    }
    int ktile = k >> 5, kin = k & 31;
    int ntile = o >> 4, n   = o & 15;
    int lane = n + ((kin >= 16) ? 16 : 0);
    int j    = kin & 15;
    dst[((size_t)(ktile * 8 + ntile) * 32 + lane) * 16 + j] = (_Float16)v;
}

// ---------------------------------------------------------------------------
// 4) gi = xh @ W_ih^T  via WMMA f16 -> f32.  [4096 x 512] x [512 x 384]
//    grid: 256 m-tiles, block 256 = 8 waves; wave w holds 3 accumulators
//    (ntiles w, w+8, w+16) so each A-fragment load feeds 3 WMMAs.
__global__ void k_gemm_ih(const _Float16* __restrict__ xh,
                          const _Float16* __restrict__ wpack,
                          float* __restrict__ gi) {
    const int mtile = blockIdx.x;
    const int wave  = threadIdx.x >> 5;
    const int lane  = threadIdx.x & 31;
    const int row0  = mtile * 16;
    const int m     = lane & 15;
    const int ko    = (lane < 16) ? 0 : 16;
    const _Float16* arow = xh + (size_t)(row0 + m) * FP;

    v8f c0 = {}, c1 = {}, c2 = {};
    #pragma unroll 4
    for (int kt = 0; kt < FP / 32; ++kt) {
        const int k0 = kt * 32;
        // prefetch B-stream 4 K-steps ahead (wrapped; speculative, droppable)
        const int ktp = (kt + 4) & (FP / 32 - 1);
        __builtin_prefetch(wpack + ((size_t)(ktp * 24 + wave) * 32 + lane) * 16, 0, 1);
        v8h a0 = *(const v8h*)(arow + k0 + ko);
        v8h a1 = *(const v8h*)(arow + k0 + 8 + ko);
        v16h a;
        #pragma unroll
        for (int e = 0; e < 8; ++e) { a[e] = a0[e]; a[e + 8] = a1[e]; }
        const _Float16* bbase = wpack + ((size_t)(kt * 24 + wave) * 32 + lane) * 16;
        v16h b0 = *(const v16h*)(bbase);
        v16h b1 = *(const v16h*)(bbase + (size_t)8 * 32 * 16);
        v16h b2 = *(const v16h*)(bbase + (size_t)16 * 32 * 16);
        c0 = __builtin_amdgcn_wmma_f32_16x16x32_f16(false, a, false, b0, (short)0, c0, false, false);
        c1 = __builtin_amdgcn_wmma_f32_16x16x32_f16(false, a, false, b1, (short)0, c1, false, false);
        c2 = __builtin_amdgcn_wmma_f32_16x16x32_f16(false, a, false, b2, (short)0, c2, false, false);
    }
    const int n     = lane & 15;
    const int mbase = (lane < 16) ? 0 : 8;
    #pragma unroll
    for (int r = 0; r < 8; ++r) {
        gi[(size_t)(row0 + mbase + r) * 384 + (wave      ) * 16 + n] = c0[r];
        gi[(size_t)(row0 + mbase + r) * 384 + (wave +  8 ) * 16 + n] = c1[r];
        gi[(size_t)(row0 + mbase + r) * 384 + (wave + 16 ) * 16 + n] = c2[r];
    }
}

// ---------------------------------------------------------------------------
// 5) GRU gates (h=0): r,z,n -> states. Writes all_states [T,B,H] (d_out) + f16 copy
__global__ void k_gates(const float* __restrict__ gi,
                        const float* __restrict__ b_ih, const float* __restrict__ b_hh,
                        float* __restrict__ states_tbh, _Float16* __restrict__ s16) {
    int idx = blockIdx.x * 256 + threadIdx.x;          // (b*1024+t)*128 + h
    if (idx >= ROWS * HH) return;
    int h   = idx & 127;
    int row = idx >> 7;                                 // b*1024 + t
    int b   = row >> 10;
    int t   = row & 1023;
    const float* g = gi + (size_t)row * 384;
    float r  = 1.0f / (1.0f + __expf(-(g[h]       + b_ih[h]       + b_hh[h])));
    float z  = 1.0f / (1.0f + __expf(-(g[128 + h] + b_ih[128 + h] + b_hh[128 + h])));
    float nn = tanhf(g[256 + h] + b_ih[256 + h] + r * b_hh[256 + h]);
    float s  = (1.0f - z) * nn;
    states_tbh[((size_t)t * BB + b) * HH + h] = s;
    s16[idx] = (_Float16)s;
}

// ---------------------------------------------------------------------------
// 6) Context gate as causal conv GEMM via WMMA. One block = (64 t's, one b).
//    8 waves, each owns a 16-wide o-tile with FOUR t-tile accumulators, so
//    every B-fragment (weight) load is reused by 4 WMMAs.  K = 8192.
__global__ void k_ctx(const _Float16* __restrict__ s16,
                      const _Float16* __restrict__ vpack,
                      const float* __restrict__ b_ctx,
                      const float* __restrict__ states_tbh,
                      float* __restrict__ gated) {
    __shared__ __align__(16) _Float16 lds[LROWS * HH]; // rows t0-63 .. t0+63 (clipped)
    const int tc = blockIdx.x;                         // 0..15
    const int b  = blockIdx.y;                         // 0..3
    const int t0 = tc * TCHUNK;

    for (int idx = threadIdx.x; idx < LROWS * HH; idx += 256) {
        int r = idx >> 7, h = idx & 127;
        int t = t0 - 63 + r; if (t < 0) t = 0;         // t0+63 <= 1023 by construction
        lds[idx] = s16[((size_t)b * TT + t) * HH + h];
    }
    __syncthreads();

    const int wave = threadIdx.x >> 5;                 // o-tile 0..7
    const int lane = threadIdx.x & 31;
    const int m    = lane & 15;
    const int ko   = (lane < 16) ? 0 : 16;

    v8f c0 = {}, c1 = {}, c2 = {}, c3 = {};
    #pragma unroll 2
    for (int kt = 0; kt < KCTX / 32; ++kt) {           // 256 steps, 4 WMMAs each
        const int k0 = kt * 32;
        const int p  = k0 >> 7;                        // tap
        const int h0 = k0 & 127;                       // h-slice base
        // prefetch next weight fragments into cache (speculative stream)
        const int ktp = (kt + 8) & (KCTX / 32 - 1);
        __builtin_prefetch(vpack + ((size_t)(ktp * 8 + wave) * 32 + lane) * 16, 0, 1);
        // one B (weight) fragment, reused for all 4 t-tiles
        v16h bf = *(const v16h*)(vpack + ((size_t)(kt * 8 + wave) * 32 + lane) * 16);
        const _Float16* lbase = lds + (m - p + 63) * HH + h0;
        #define CTX_STEP(cc, ttt)                                                   \
        {                                                                           \
            const _Float16* lrow = lbase + (ttt) * 16 * HH;                         \
            v8h a0 = *(const v8h*)(lrow + ko);                                      \
            v8h a1 = *(const v8h*)(lrow + 8 + ko);                                  \
            v16h a;                                                                 \
            _Pragma("unroll")                                                       \
            for (int e = 0; e < 8; ++e) { a[e] = a0[e]; a[e + 8] = a1[e]; }         \
            cc = __builtin_amdgcn_wmma_f32_16x16x32_f16(false, a, false, bf,        \
                                                        (short)0, cc, false, false);\
        }
        CTX_STEP(c0, 0)
        CTX_STEP(c1, 1)
        CTX_STEP(c2, 2)
        CTX_STEP(c3, 3)
        #undef CTX_STEP
    }

    const int n     = lane & 15;
    const int mbase = (lane < 16) ? 0 : 8;
    const int o     = wave * 16 + n;
    const float bc  = b_ctx[o];
    v8f cs[4] = {c0, c1, c2, c3};
    #pragma unroll
    for (int tt = 0; tt < 4; ++tt) {
        #pragma unroll
        for (int r = 0; r < 8; ++r) {
            int   t = t0 + tt * 16 + mbase + r;
            float y = cs[tt][r] + bc;
            float g = 1.0f / (1.0f + __expf(-y));
            float s = states_tbh[((size_t)t * BB + b) * HH + o];
            gated[((size_t)b * TT + t) * HH + o] = s * g;
        }
    }
}

// ---------------------------------------------------------------------------
// 7) probs = clip(sigmoid(gated @ W_fc^T + b_fc))  [4096 x 128] x [128 x 64]
__global__ void k_fc(const float* __restrict__ gated,
                     const float* __restrict__ wfc, const float* __restrict__ bfc,
                     float* __restrict__ probs) {
    __shared__ float row[HH];
    const int bt = blockIdx.x;                         // b*1024 + t
    const int p  = threadIdx.x;                        // 0..63
    row[p]      = gated[(size_t)bt * HH + p];
    row[p + 64] = gated[(size_t)bt * HH + p + 64];
    __syncthreads();
    float acc = bfc[p];
    #pragma unroll 8
    for (int h = 0; h < HH; ++h) acc += row[h] * wfc[p * HH + h];
    float pr = 1.0f / (1.0f + __expf(-acc));
    pr = fminf(fmaxf(pr, 0.001f), 0.999f);
    probs[(size_t)bt * PP + p] = pr;
}

// ---------------------------------------------------------------------------
extern "C" void kernel_launch(void* const* d_in, const int* in_sizes, int n_in,
                              void* d_out, int out_size, void* d_ws, size_t ws_size,
                              hipStream_t stream) {
    const float* x     = (const float*)d_in[0];
    const float* W_ih  = (const float*)d_in[1];
    // d_in[2] = W_hh : unused (hidden state is always zero in the reference)
    const float* b_ih  = (const float*)d_in[3];
    const float* b_hh  = (const float*)d_in[4];
    const float* W_ctx = (const float*)d_in[5];
    const float* b_ctx = (const float*)d_in[6];
    const float* W_fc  = (const float*)d_in[7];
    const float* b_fc  = (const float*)d_in[8];

    float* out        = (float*)d_out;
    float* probs      = out;                            // [B,T,P] = 262144
    float* states_tbh = out + (size_t)BB * TT * PP;     // [T,B,H] = 524288

    char* ws = (char*)d_ws;
    _Float16* xh    = (_Float16*)(ws + 0);              //  4,194,304 B
    _Float16* wihp  = (_Float16*)(ws + 4194304);        //    393,216 B
    float*    gi    = (float*)   (ws + 4587520);        //  6,291,456 B
    _Float16* s16   = (_Float16*)(ws + 10878976);       //  1,048,576 B
    _Float16* vpack = (_Float16*)(ws + 11927552);       //  2,097,152 B
    float*    gated = (float*)   (ws + 14024704);       //  2,097,152 B -> 16.1 MB

    k_convert_x<<<(ROWS * FP) / 256, 256, 0, stream>>>(x, xh);
    k_pack_wih <<<(384 * FP) / 256, 256, 0, stream>>>(W_ih, wihp);
    k_pack_vctx<<<(HH * KCTX) / 256, 256, 0, stream>>>(W_ctx, vpack);

    k_gemm_ih<<<ROWS / 16, 256, 0, stream>>>(xh, wihp, gi);
    k_gates  <<<(ROWS * HH) / 256, 256, 0, stream>>>(gi, b_ih, b_hh, states_tbh, s16);

    k_ctx<<<dim3(TT / TCHUNK, BB), 256, 0, stream>>>(s16, vpack, b_ctx, states_tbh, gated);

    k_fc<<<ROWS, PP, 0, stream>>>(gated, W_fc, b_fc, probs);
}
